// SO3EquivariantConv_56573309223683
// MI455X (gfx1250) — compile-verified, matrix operand order checked by reading, not run on hardware
//
#include <hip/hip_runtime.h>
#include <math.h>

typedef __attribute__((ext_vector_type(16))) _Float16 v16h;
typedef __attribute__((ext_vector_type(8)))  _Float16 v8h;
typedef __attribute__((ext_vector_type(2)))  __fp16   v2fp16;  // cvt_pkrtz return type
typedef __attribute__((ext_vector_type(8)))  float    v8f;

#define C_IN  64
#define C_OUT 32
#define NSH   9
#define HDIM  32
#define KPAD  96   // 73 padded to 3 x 32; layout: [feat 0..63 | SH 64..72 | zero 73..95]
#define WPB   8    // waves per block

// ---------- helpers ----------

__device__ __forceinline__ void wave_lds_fence() {
  // LDS ops are in-order within a wave; this blocks compiler reordering of the
  // cross-lane LDS stores vs the dependent fragment loads.
  asm volatile("s_wait_dscnt 0" ::: "memory");
}

__device__ __forceinline__ v16h load_frag16(const _Float16* base) {
  // per ISA 7.12.2: lane picks K {kb..kb+7} and {kb+16..kb+23}
  v8h lo = *(const v8h*)(base);
  v8h hi = *(const v8h*)(base + 16);
  return __builtin_shufflevector(lo, hi, 0,1,2,3,4,5,6,7,8,9,10,11,12,13,14,15);
}

__device__ __forceinline__ v8f splat8(float x) {
  v8f r;
#pragma unroll
  for (int i = 0; i < 8; ++i) r[i] = x;
  return r;
}

__device__ __forceinline__ float silu_f(float x) {
  // x * sigmoid(x): v_exp + v_add + v_rcp + v_mul (no IEEE div chain)
  float e = __expf(-x);
  return x * __builtin_amdgcn_rcpf(1.0f + e);
}

// ---------- weight prep: fp32 -> f16 col-major, permute W1 rows, reduce W3 ----------

__global__ void so3_prep(const float* __restrict__ W1, const float* __restrict__ W2,
                         const float* __restrict__ W3, const float* __restrict__ b3,
                         _Float16* __restrict__ w1h, _Float16* __restrict__ w2h,
                         _Float16* __restrict__ w3h, float* __restrict__ b3r) {
  int tid = threadIdx.x;
  // W1 [73,32] row-major -> col-major [n][k] with k permuted:
  // new k<64 = old row 9+k (features), new 64..72 = old rows 0..8 (SH), pad 0.
  for (int i = tid; i < HDIM * KPAD; i += 256) {
    int n = i / KPAD, k = i % KPAD;
    float v = 0.0f;
    if (k < C_IN)                 v = W1[(NSH + k) * HDIM + n];
    else if (k < C_IN + NSH)      v = W1[(k - C_IN) * HDIM + n];
    w1h[i] = (_Float16)v;
  }
  for (int i = tid; i < HDIM * HDIM; i += 256) {
    int n = i / HDIM, k = i % HDIM;
    w2h[i] = (_Float16)W2[k * HDIM + n];
  }
  // W3 reduced over 9 SH groups -> [k][c] col-major
  for (int i = tid; i < HDIM * C_OUT; i += 256) {
    int n = i / HDIM, k = i % HDIM;
    float s = 0.0f;
#pragma unroll
    for (int sh = 0; sh < NSH; ++sh) s += W3[k * (NSH * C_OUT) + sh * C_OUT + n];
    w3h[i] = (_Float16)s;
  }
  for (int i = tid; i < C_OUT; i += 256) {
    float s = 0.0f;
#pragma unroll
    for (int sh = 0; sh < NSH; ++sh) s += b3[sh * C_OUT + i];
    b3r[i] = s;
  }
}

// ---------- main fused kernel ----------

__launch_bounds__(32 * WPB, 1)
__global__ void so3_conv(const float* __restrict__ feats,
                         const float* __restrict__ pos,
                         const float* __restrict__ b1,
                         const float* __restrict__ b2,
                         const int*   __restrict__ eidx,   // [2,E] int32
                         const _Float16* __restrict__ w1h,
                         const _Float16* __restrict__ w2h,
                         const _Float16* __restrict__ w3h,
                         const float* __restrict__ b3r,
                         float* __restrict__ out,
                         int E, int numTiles) {
  __shared__ _Float16 sA[WPB][16 * KPAD];
  __shared__ _Float16 sH[WPB][16 * HDIM];

  const int lane  = threadIdx.x & 31;
  const int wave  = threadIdx.x >> 5;
  const int l15   = lane & 15;
  const int hi16  = (lane >> 4) & 1;
  const int koff  = hi16 * 8;   // fragment K sub-offset (ISA 7.12.2)
  const int mbase = hi16 * 8;   // C/D row base

  // --- preload weight B-fragments into VGPRs (live across tile loop) ---
  v16h w1f[3][2], w2f[2], w3f[2];
#pragma unroll
  for (int kc = 0; kc < 3; ++kc)
#pragma unroll
    for (int nh = 0; nh < 2; ++nh)
      w1f[kc][nh] = load_frag16(w1h + (nh * 16 + l15) * KPAD + kc * 32 + koff);
#pragma unroll
  for (int nh = 0; nh < 2; ++nh) {
    w2f[nh] = load_frag16(w2h + (nh * 16 + l15) * HDIM + koff);
    w3f[nh] = load_frag16(w3h + (nh * 16 + l15) * HDIM + koff);
  }
  float b1v[2], b2v[2], b3v[2];
#pragma unroll
  for (int nh = 0; nh < 2; ++nh) {
    b1v[nh] = b1[nh * 16 + l15];
    b2v[nh] = b2[nh * 16 + l15];
    b3v[nh] = b3r[nh * 16 + l15];
  }

  _Float16* A   = sA[wave];
  _Float16* Hs  = sH[wave];
  uint32_t* A32 = (uint32_t*)A;

  // zero A once: pad columns 73..95 stay zero for all tiles
  for (int i = lane; i < 16 * KPAD / 2; i += 32) A32[i] = 0u;

  const int waveGlobal = blockIdx.x * WPB + wave;
  const int waveCount  = gridDim.x * WPB;

  for (int t = waveGlobal; t < numTiles; t += waveCount) {
    const int e0 = t * 16;
    const int e  = e0 + l15;          // lanes m and m+16 both hold edge m

    int rIdx = -1, cIdx = 0;
    if (e < E) {
      rIdx = eidx[e];
      cIdx = eidx[E + e];
    }

    // ---- phase 1: spherical harmonics (lanes 0-15 write rows 0-15)
    if (lane < 16) {
      float dx = 0.0f, dy = 0.0f, dz = 0.0f;
      if (e < E) {
        dx = pos[3 * rIdx + 0] - pos[3 * cIdx + 0];
        dy = pos[3 * rIdx + 1] - pos[3 * cIdx + 1];
        dz = pos[3 * rIdx + 2] - pos[3 * cIdx + 2];
      }
      float d2  = dx * dx + dy * dy + dz * dz;
      float inv = fminf(__builtin_amdgcn_rsqf(d2), 1e8f);
      float x = dx * inv, y = dy * inv, z = dz * inv;
      const float c00 = 0.28209479177387814f;
      const float c1  = 0.3454941494713355f;
      const float c1z = 0.4886025119029199f;
      const float c2  = 0.38627420202318958f;
      const float c2z = 0.7725484040463792f;
      const float c20 = 0.31539156525252005f;
      float xx = x * x, yy = y * y, zz = z * z;
      float sh4 = c2 * (xx - yy);
      float sh7 = c2z * z * x;
      v8h shv;
      shv[0] = (_Float16)(c00);
      shv[1] = (_Float16)(-c1 * x);
      shv[2] = (_Float16)(c1z * z);
      shv[3] = (_Float16)(c1 * x);
      shv[4] = (_Float16)(sh4);
      shv[5] = (_Float16)(-sh7);
      shv[6] = (_Float16)(c20 * (2.0f * zz - xx - yy));
      shv[7] = (_Float16)(sh7);
      _Float16* rowp = A + l15 * KPAD;
      *(v8h*)(rowp + C_IN) = shv;          // 16B-aligned (l15*192 + 128)
      rowp[C_IN + 8] = (_Float16)(sh4);
    }

    // ---- phase 2: feature gather, one float2 per lane per edge (256B coalesced)
#pragma unroll
    for (int ee = 0; ee < 16; ++ee) {
      int src = __builtin_amdgcn_readlane(cIdx, ee);   // uniform lane index
      const float2 fv = *(const float2*)(feats + (size_t)src * C_IN + 2 * lane);
      union { v2fp16 h; uint32_t u; } cv;
      cv.h = __builtin_amdgcn_cvt_pkrtz(fv.x, fv.y);
      A32[ee * (KPAD / 2) + lane] = cv.u;
    }
    wave_lds_fence();

    // ---- layer 1: [16,96] @ [96,32] + b1, silu
    v8f acc[2] = { splat8(b1v[0]), splat8(b1v[1]) };
#pragma unroll
    for (int kc = 0; kc < 3; ++kc) {
      v16h a = load_frag16(A + l15 * KPAD + kc * 32 + koff);
      acc[0] = __builtin_amdgcn_wmma_f32_16x16x32_f16(false, a, false, w1f[kc][0],
                                                      (short)0, acc[0], false, false);
      acc[1] = __builtin_amdgcn_wmma_f32_16x16x32_f16(false, a, false, w1f[kc][1],
                                                      (short)0, acc[1], false, false);
    }
#pragma unroll
    for (int nh = 0; nh < 2; ++nh) {
      int n = nh * 16 + l15;
#pragma unroll
      for (int r = 0; r < 8; ++r)
        Hs[(mbase + r) * HDIM + n] = (_Float16)silu_f(acc[nh][r]);
    }
    wave_lds_fence();

    // ---- layer 2: [16,32] @ [32,32] + b2, silu
    {
      v16h a = load_frag16(Hs + l15 * HDIM + koff);
      v8f a2[2] = { splat8(b2v[0]), splat8(b2v[1]) };
      a2[0] = __builtin_amdgcn_wmma_f32_16x16x32_f16(false, a, false, w2f[0],
                                                     (short)0, a2[0], false, false);
      a2[1] = __builtin_amdgcn_wmma_f32_16x16x32_f16(false, a, false, w2f[1],
                                                     (short)0, a2[1], false, false);
#pragma unroll
      for (int nh = 0; nh < 2; ++nh) {
        int n = nh * 16 + l15;
#pragma unroll
        for (int r = 0; r < 8; ++r)
          Hs[(mbase + r) * HDIM + n] = (_Float16)silu_f(a2[nh][r]);
      }
    }
    wave_lds_fence();

    // ---- layer 3: [16,32] @ [32,32] + b3r, scatter-add to out[row]
    {
      v16h a = load_frag16(Hs + l15 * HDIM + koff);
      v8f a3[2] = { splat8(b3v[0]), splat8(b3v[1]) };
      a3[0] = __builtin_amdgcn_wmma_f32_16x16x32_f16(false, a, false, w3f[0],
                                                     (short)0, a3[0], false, false);
      a3[1] = __builtin_amdgcn_wmma_f32_16x16x32_f16(false, a, false, w3f[1],
                                                     (short)0, a3[1], false, false);
#pragma unroll
      for (int r = 0; r < 8; ++r) {
        int dst = __shfl(rIdx, mbase + r, 32);   // per-half-wave src lane -> bpermute
        if (dst >= 0) {
#pragma unroll
          for (int nh = 0; nh < 2; ++nh)
            __hip_atomic_fetch_add(out + dst * C_OUT + nh * 16 + l15, a3[nh][r],
                                   __ATOMIC_RELAXED, __HIP_MEMORY_SCOPE_AGENT);
        }
      }
    }
    // no trailing fence needed: next-tile A stores don't alias Hs reads, and
    // A-vs-A ordering is covered by the phase fences + in-order LDS per wave.
  }
}

// ---------- launcher ----------

extern "C" void kernel_launch(void* const* d_in, const int* in_sizes, int n_in,
                              void* d_out, int out_size, void* d_ws, size_t ws_size,
                              hipStream_t stream) {
  const float* feats = (const float*)d_in[0];
  const float* pos   = (const float*)d_in[1];
  const float* W1    = (const float*)d_in[2];
  const float* b1    = (const float*)d_in[3];
  const float* W2    = (const float*)d_in[4];
  const float* b2    = (const float*)d_in[5];
  const float* W3    = (const float*)d_in[6];
  const float* b3    = (const float*)d_in[7];
  const int*   eidx  = (const int*)d_in[8];

  const int E = in_sizes[8] / 2;

  _Float16* w1h = (_Float16*)d_ws;                 // 32*96 halfs
  _Float16* w2h = w1h + HDIM * KPAD;               // 32*32
  _Float16* w3h = w2h + HDIM * HDIM;               // 32*32
  float*    b3r = (float*)(w3h + HDIM * HDIM);     // 32 floats

  (void)hipMemsetAsync(d_out, 0, (size_t)out_size * sizeof(float), stream);

  so3_prep<<<1, 256, 0, stream>>>(W1, W2, W3, b3, w1h, w2h, w3h, b3r);

  const int numTiles = (E + 15) / 16;
  int blocks = (numTiles + WPB - 1) / WPB;
  if (blocks > 1024) blocks = 1024;
  if (blocks < 1) blocks = 1;

  so3_conv<<<blocks, 32 * WPB, 0, stream>>>(
      feats, pos, b1, b2, eidx, w1h, w2h, w3h, b3r, (float*)d_out, E, numTiles);
}
